// VecAttGRUCell_13134009991515
// MI455X (gfx1250) — matrix operand order repaired and checked by reading, not compiled, not run on hardware
//
#include <hip/hip_runtime.h>
#include <hip/hip_bf16.h>
#include <math.h>

typedef __attribute__((ext_vector_type(16))) _Float16 v16h;
typedef __attribute__((ext_vector_type(8)))  _Float16 v8h;
typedef __attribute__((ext_vector_type(8)))  float    v8f;

#define B_ROWS 16384
#define UNITS  1024
#define KDIM   2048   // D_IN + UNITS
#define N_GATE 2048   // 2*UNITS
#define N_CAND 1024

// ---------------------------------------------------------------------------
// fp32 -> f16 conversion (activations)
// ---------------------------------------------------------------------------
__global__ void cvt_f32_f16(const float* __restrict__ src,
                            _Float16* __restrict__ dst, int n) {
  int i = blockIdx.x * blockDim.x + threadIdx.x;
  if (i < n) dst[i] = (_Float16)src[i];
}

// ---------------------------------------------------------------------------
// Pre-shuffle row-major fp32 weights (K x N) into per-tile f16 B-fragment
// layout.  Tile = 32(K) x 16(N).  Storage:
//   Wsh[ nt * (K/32) * 512  +  kt * 512  +  lane * 16  +  e ]
//     = W[ kt*32 + (e/8)*16 + (lane/16)*8 + (e%8) ][ nt*16 + (lane%16) ]
// so each lane of the GEMM wave reads its 16 halves as two contiguous b128s.
// ---------------------------------------------------------------------------
__global__ void shuffle_w(const float* __restrict__ W,
                          _Float16* __restrict__ Wsh, int K, int N) {
  int idx = blockIdx.x * blockDim.x + threadIdx.x;
  int total = K * N;
  if (idx >= total) return;
  int e    = idx & 15;
  int lane = (idx >> 4) & 31;
  int ktiles = K >> 5;
  int kt   = (idx >> 9) % ktiles;
  int nt   = idx / (512 * ktiles);
  int k = kt * 32 + ((e >> 3) << 4) + ((lane >> 4) << 3) + (e & 7);
  int n = nt * 16 + (lane & 15);
  Wsh[idx] = (_Float16)W[(size_t)k * N + n];
}

// ---------------------------------------------------------------------------
// Fused GEMM + AUGRU epilogue.
//   mode 0 (gates):  v = sigmoid(acc + b_gate[n])
//       n <  1024 : rh16[m][n]      = (f16)(v * h[m][n])
//       n >= 1024 : u16[m][n-1024]  = (f16)((1 - att[m]) * v)
//   mode 1 (cand):   c = tanh(acc + b_cand[n]); u = u16[m][n]
//       out[m][n] = u*h[m][n] + (1-u)*c
// Block: 256 threads (8 waves), tile 128M x 256N; wave tile 64M x 64N
// (4x4 subtiles -> 16 WMMAs per K-step, 16 b128 loads: 1.0 loads/WMMA).
// ---------------------------------------------------------------------------
__launch_bounds__(256)
__global__ void gru_gemm(const _Float16* __restrict__ Alo,   // B x 1024 f16
                         const _Float16* __restrict__ Ahi,   // B x 1024 f16
                         const _Float16* __restrict__ Wsh,   // shuffled f16
                         const float* __restrict__ bias,
                         const float* __restrict__ h,        // B x 1024 f32
                         const float* __restrict__ att,      // B x 1
                         const _Float16* __restrict__ u_in,  // mode 1
                         _Float16* __restrict__ rh_out,      // mode 0
                         _Float16* __restrict__ u_out,       // mode 0
                         float* __restrict__ out,            // mode 1
                         int mode) {
  const int lane = threadIdx.x & 31;
  const int wave = threadIdx.x >> 5;
  const int wm = wave & 1;                    // 0..1
  const int wn = wave >> 1;                   // 0..3
  const int m0 = blockIdx.y * 128 + wm * 64;  // wave M base (4 subtiles)
  const int n0 = blockIdx.x * 256 + wn * 64;  // wave N base (4 subtiles)

  const int ktiles = KDIM >> 5;               // 64
  const int mrow = lane & 15;
  const int kg8  = (lane >> 4) << 3;          // 0 or 8 (A K sub-chunk)

  v8f acc[4][4];
#pragma unroll
  for (int mt = 0; mt < 4; ++mt)
#pragma unroll
    for (int nt = 0; nt < 4; ++nt)
#pragma unroll
      for (int r = 0; r < 8; ++r) acc[mt][nt][r] = 0.0f;

  // K loop split in two halves: k < 1024 reads Alo, k >= 1024 reads Ahi.
  for (int half = 0; half < 2; ++half) {
    const _Float16* __restrict__ A = half ? Ahi : Alo;
    for (int kt2 = 0; kt2 < (ktiles >> 1); ++kt2) {
      const int kt = half * (ktiles >> 1) + kt2;
      const int kl = kt2 * 32;

      v16h afrag[4];
#pragma unroll
      for (int mt = 0; mt < 4; ++mt) {
        const _Float16* ap =
            A + (size_t)(m0 + mt * 16 + mrow) * UNITS + kl + kg8;
        v8h lo = *(const v8h*)(ap);
        v8h hi = *(const v8h*)(ap + 16);
        afrag[mt] = __builtin_shufflevector(
            lo, hi, 0, 1, 2, 3, 4, 5, 6, 7, 8, 9, 10, 11, 12, 13, 14, 15);
      }

#pragma unroll
      for (int nt = 0; nt < 4; ++nt) {
        const int ntg = (n0 >> 4) + nt;
        const _Float16* bp =
            Wsh + ((size_t)ntg * ktiles + kt) * 512 + lane * 16;
        v8h lo = *(const v8h*)(bp);
        v8h hi = *(const v8h*)(bp + 8);
        v16h bfrag = __builtin_shufflevector(
            lo, hi, 0, 1, 2, 3, 4, 5, 6, 7, 8, 9, 10, 11, 12, 13, 14, 15);
#pragma unroll
        for (int mt = 0; mt < 4; ++mt) {
          acc[mt][nt] = __builtin_amdgcn_wmma_f32_16x16x32_f16(
              false, afrag[mt], false, bfrag, (short)0, acc[mt][nt],
              false, false);
        }
      }
    }
  }

  // ---------------- epilogue ----------------
  const int colbase = n0 + (lane & 15);
  const int rowoff  = (lane >> 4) * 8;
#pragma unroll
  for (int mt = 0; mt < 4; ++mt) {
#pragma unroll
    for (int nt = 0; nt < 4; ++nt) {
      const int n = colbase + nt * 16;
      const float bn = bias[n];
#pragma unroll
      for (int r = 0; r < 8; ++r) {
        const int m = m0 + mt * 16 + rowoff + r;
        const float v = acc[mt][nt][r] + bn;
        if (mode == 0) {
          const float g = 1.0f / (1.0f + __expf(-v));
          if (n < UNITS) {  // reset gate -> r*h (f16) feeds candidate GEMM
            rh_out[(size_t)m * UNITS + n] =
                (_Float16)(g * h[(size_t)m * UNITS + n]);
          } else {          // update gate, attention-modified
            const float u = (1.0f - att[m]) * g;
            u_out[(size_t)m * UNITS + (n - UNITS)] = (_Float16)u;
          }
        } else {
          const float c = tanhf(v);
          const float u = (float)u_in[(size_t)m * UNITS + n];
          const float hv = h[(size_t)m * UNITS + n];
          out[(size_t)m * UNITS + n] = u * hv + (1.0f - u) * c;
        }
      }
    }
  }
}

// ---------------------------------------------------------------------------
extern "C" void kernel_launch(void* const* d_in, const int* in_sizes, int n_in,
                              void* d_out, int out_size, void* d_ws,
                              size_t ws_size, hipStream_t stream) {
  const float* x   = (const float*)d_in[0];  // (B, 1024)
  const float* att = (const float*)d_in[1];  // (B, 1)
  const float* h   = (const float*)d_in[2];  // (B, 1024)
  const float* Wg  = (const float*)d_in[3];  // (2048, 2048)
  const float* bg  = (const float*)d_in[4];  // (2048,)
  const float* Wc  = (const float*)d_in[5];  // (2048, 1024)
  const float* bc  = (const float*)d_in[6];  // (1024,)
  float* out = (float*)d_out;

  char* ws = (char*)d_ws;
  const size_t act_bytes = (size_t)B_ROWS * UNITS * sizeof(_Float16);  // 33.5MB
  _Float16* x16  = (_Float16*)(ws + 0 * act_bytes);
  _Float16* h16  = (_Float16*)(ws + 1 * act_bytes);
  _Float16* rh16 = (_Float16*)(ws + 2 * act_bytes);
  _Float16* u16  = (_Float16*)(ws + 3 * act_bytes);
  _Float16* Wgs  = (_Float16*)(ws + 4 * act_bytes);
  _Float16* Wcs  = (_Float16*)((char*)Wgs + (size_t)KDIM * N_GATE * sizeof(_Float16));

  const int nconv = B_ROWS * UNITS;
  cvt_f32_f16<<<(nconv + 255) / 256, 256, 0, stream>>>(x, x16, nconv);
  cvt_f32_f16<<<(nconv + 255) / 256, 256, 0, stream>>>(h, h16, nconv);

  const int wg_elems = KDIM * N_GATE;
  const int wc_elems = KDIM * N_CAND;
  shuffle_w<<<(wg_elems + 255) / 256, 256, 0, stream>>>(Wg, Wgs, KDIM, N_GATE);
  shuffle_w<<<(wc_elems + 255) / 256, 256, 0, stream>>>(Wc, Wcs, KDIM, N_CAND);

  // GEMM1: gates (N = 2048), fused sigmoid + r*h + (1-att)*u
  dim3 g1(N_GATE / 256, B_ROWS / 128);  // (8, 128)
  gru_gemm<<<g1, 256, 0, stream>>>(x16, h16, Wgs, bg, h, att,
                                   (const _Float16*)nullptr, rh16, u16,
                                   (float*)nullptr, 0);

  // GEMM2: candidate (N = 1024), fused tanh + AUGRU blend -> out
  dim3 g2(N_CAND / 256, B_ROWS / 128);  // (4, 128)
  gru_gemm<<<g2, 256, 0, stream>>>(x16, rh16, Wcs, bc, h, att, u16,
                                   (_Float16*)nullptr, (_Float16*)nullptr,
                                   out, 1);
}
